// MHSA_36404142801396
// MI455X (gfx1250) — compile-verified, matrix operand order checked by reading, not run on hardware
//
#include <hip/hip_runtime.h>

typedef __attribute__((ext_vector_type(16))) _Float16 v16h;
typedef __attribute__((ext_vector_type(8)))  float    v8f;
typedef __attribute__((ext_vector_type(4)))  unsigned int v4u;
typedef __attribute__((ext_vector_type(8)))  int      v8i;
typedef __attribute__((ext_vector_type(4)))  int      v4i;

#define NH    8
#define HD    64
#define EMB   512
#define BB    4
#define SS    2048
#define MTOT  (BB*SS)          /* 8192 */
#define NQKV  (3*NH*HD)        /* 1536 */
#define QSZ   ((size_t)BB*NH*SS*HD)  /* 4194304 elements per q/k/v */

union HFrag { v16h v; unsigned int u[8]; };
union CAcc  { v8f  v; float f[8]; };

// ISA 16-bit A/B fragment k index for vector pair p (elements 2p,2p+1), half = lane/16
__device__ __forceinline__ int kpair(int p, int hi) {
  return ((p & 4) << 2) + (hi << 3) + ((p & 3) << 1);
}

// TDM 2-D tile load: global (fp16 tensor, row stride stride0 elements) -> LDS,
// with row padding so the LDS image has a bank-conflict-free stride.
// Descriptor packing per CDNA5 ISA section 8.3/8.4 (D# group 0/1), 2-D tile:
//   group0: count=1 | lds_addr | global_addr[56:0] | type=2
//   group1: data_size=1(2B), pad_enable, pad_interval, pad_amount,
//           tensor_dim0/1, tile_dim0/1, tensor_dim0_stride
// Toolchain uses the 6-arg builtin: (v4u, v8i, v4i, v4i, v8i, i32 cpol).
__device__ __forceinline__ void tdm_load_2d(
    unsigned int ldsAddr, const _Float16* gptr,
    unsigned int tensorDim0, unsigned int tensorDim1,
    unsigned int tileDim0, unsigned int tileDim1,
    unsigned int stride0, unsigned int padInterval, unsigned int padAmount)
{
  unsigned long long ga = (unsigned long long)(size_t)gptr;
  v4u g0;
  g0[0] = 1u;                                       // count=1, user descriptor
  g0[1] = ldsAddr;                                  // LDS byte address
  g0[2] = (unsigned int)ga;                         // global_addr[31:0]
  g0[3] = (unsigned int)(ga >> 32) | (2u << 30);    // global_addr[56:32] | type=2
  v8i g1;
  g1[0] = (int)((1u << 16) | (1u << 20) | (padInterval << 22) | (padAmount << 25));
  g1[1] = (int)((tensorDim0 & 0xffffu) << 16);
  g1[2] = (int)((tensorDim0 >> 16) | ((tensorDim1 & 0xffffu) << 16));
  g1[3] = (int)((tensorDim1 >> 16) | (tileDim0 << 16));
  g1[4] = (int)(tileDim1 & 0xffffu);                // tile_dim2 = 0 (2-D)
  g1[5] = (int)stride0;                             // tensor_dim0_stride[31:0]
  g1[6] = 0;                                        // stride hi / dim1_stride lo
  g1[7] = 0;
  v4i g2 = {0, 0, 0, 0};
  v4i g3 = {0, 0, 0, 0};
  v8i g4 = {0, 0, 0, 0, 0, 0, 0, 0};
  __builtin_amdgcn_tensor_load_to_lds(g0, g1, g2, g3, g4, 0);
}

__global__ void cvt_f32_to_f16(const float* __restrict__ src,
                               _Float16* __restrict__ dst, int n) {
  int i = blockIdx.x * 256 + threadIdx.x;
  if (i < n) dst[i] = (_Float16)src[i];
}

// dst[n*K + k] = (f16) src[k*N + n]  -- pre-transpose weights so GEMM B tiles
// are plain strided 2-D copies (TDM cannot transpose).
__global__ void cvt_transpose_f32_to_f16(const float* __restrict__ src,
                                         _Float16* __restrict__ dst,
                                         int K, int N) {
  int i = blockIdx.x * 256 + threadIdx.x;
  if (i < K * N) {
    int n = i / K, kk = i % K;
    dst[i] = (_Float16)src[(size_t)kk * N + n];
  }
}

// C = A[M,K] @ Bt[N,K]^T + bias, fp16 in / f32 accum via v_wmma_f32_16x16x32_f16.
// Block = 256 threads = 8 wave32s; block tile 128(M) x 128(N); K step 32;
// per-wave tile 32x64 (2 A-frags x 4 B-frags = 8 WMMA / K-step).
// All staging via TDM (tensor_load_to_lds), double-buffered, wave 0 issues.
// mode 0: scatter q/k (seq-major) and v (pre-transposed [B,H,D,S]); mode 1: f32 out.
__global__ __launch_bounds__(256) void gemm_f16_wmma(
    const _Float16* __restrict__ A,
    const _Float16* __restrict__ Bt,   // [N][K]
    const float* __restrict__ bias,
    int Kdim, int N, int mode,
    _Float16* __restrict__ qkv_dst,
    float* __restrict__ out_f32)
{
  __shared__ __align__(16) _Float16 ldsA[2][128 * 34];  // [row][k], stride 34 (pad)
  __shared__ __align__(16) _Float16 ldsB[2][128 * 34];  // [n][k],  stride 34 (pad)

  const int tid   = threadIdx.x;
  const int w     = tid >> 5;
  const int lane  = tid & 31;
  const int hi    = lane >> 4;
  const int ln    = lane & 15;
  const int waveM = w & 3;                 // 4 strips of 32 rows
  const int waveN = w >> 2;                // 2 strips of 64 cols
  const int rowBlock = blockIdx.y * 128;
  const int nBlock   = blockIdx.x * 128;
  const int NSTEP    = Kdim >> 5;

  v8f zero = {0.f,0.f,0.f,0.f,0.f,0.f,0.f,0.f};
  CAcc acc0[4], acc1[4];
#pragma unroll
  for (int t = 0; t < 4; ++t) { acc0[t].v = zero; acc1[t].v = zero; }

  auto issueTiles = [&](int buf, int k0) {
    // A tile: 128 rows x 32 k, rows padded 16 dwords + 1 (interval code 3)
    tdm_load_2d((unsigned int)(size_t)&ldsA[buf][0],
                A + (size_t)rowBlock * Kdim + k0,
                (unsigned)Kdim, (unsigned)MTOT, 32, 128, (unsigned)Kdim, 3, 0);
    // B tile: 128 n-rows x 32 k from pre-transposed weights
    tdm_load_2d((unsigned int)(size_t)&ldsB[buf][0],
                Bt + (size_t)nBlock * Kdim + k0,
                (unsigned)Kdim, (unsigned)N, 32, 128, (unsigned)Kdim, 3, 0);
  };

  if (w == 0) {
    issueTiles(0, 0);
    __builtin_amdgcn_s_wait_tensorcnt(0);
  }
  __syncthreads();

  for (int step = 0; step < NSTEP; ++step) {
    const int cur = step & 1, nxt = cur ^ 1;
    const bool hasNext = (step + 1) < NSTEP;
    if (w == 0 && hasNext) issueTiles(nxt, (step + 1) * 32);

    HFrag a0, a1;
    const _Float16* ar0 = &ldsA[cur][(waveM * 32 + ln) * 34];
    const _Float16* ar1 = ar0 + 16 * 34;
#pragma unroll
    for (int p = 0; p < 8; ++p) {
      a0.u[p] = *(const unsigned int*)(ar0 + kpair(p, hi));
      a1.u[p] = *(const unsigned int*)(ar1 + kpair(p, hi));
    }
#pragma unroll
    for (int t = 0; t < 4; ++t) {
      HFrag b;
      const _Float16* br = &ldsB[cur][(waveN * 64 + t * 16 + ln) * 34];
#pragma unroll
      for (int p = 0; p < 8; ++p) b.u[p] = *(const unsigned int*)(br + kpair(p, hi));
      acc0[t].v = __builtin_amdgcn_wmma_f32_16x16x32_f16(
          false, a0.v, false, b.v, (short)0, acc0[t].v, false, false);
      acc1[t].v = __builtin_amdgcn_wmma_f32_16x16x32_f16(
          false, a1.v, false, b.v, (short)0, acc1[t].v, false, false);
    }

    if (w == 0 && hasNext) __builtin_amdgcn_s_wait_tensorcnt(0);
    __syncthreads();
  }

#pragma unroll
  for (int t = 0; t < 4; ++t) {
    int col = nBlock + waveN * 64 + t * 16 + ln;
    float bv = bias[col];
#pragma unroll
    for (int mi = 0; mi < 2; ++mi) {
      CAcc* acc = mi ? acc1 : acc0;
#pragma unroll
      for (int r = 0; r < 8; ++r) {
        int row = rowBlock + waveM * 32 + mi * 16 + r + (hi << 3);
        float vv = acc[t].f[r] + bv;
        if (mode == 0) {
          int which = col >> 9;         // 0=q 1=k 2=v
          int h = (col >> 6) & 7;
          int d = col & 63;
          int b = row >> 11;            // / 2048
          int s = row & 2047;
          size_t idx;
          if (which == 2) {             // v stored transposed: [B,H,D,S]
            idx = 2 * QSZ + ((size_t)(b * NH + h) * HD + d) * SS + s;
          } else {
            if (which == 0) vv *= 0.125f;  // D^-0.5 folded into q
            idx = (size_t)which * QSZ + ((size_t)(b * NH + h) * SS + s) * HD + d;
          }
          qkv_dst[idx] = (_Float16)vv;
        } else {
          out_f32[(size_t)row * N + col] = vv;
        }
      }
    }
  }
}

// Flash attention: block = 256 threads (8 waves), 128 q rows per block sharing
// one K/V tile; K ([kv][d]) and pre-transposed V ([d][kv]) staged by TDM with
// double buffering; online softmax; QK^T and P*V via WMMA.
__global__ __launch_bounds__(256) void flash_attn_wmma(
    const _Float16* __restrict__ q,    // [B,H,S,D]
    const _Float16* __restrict__ k,    // [B,H,S,D]
    const _Float16* __restrict__ vt,   // [B,H,D,S]
    _Float16* __restrict__ o_out)      // [B,S,H*D]
{
  __shared__ __align__(16) _Float16 ldsK[2][64 * 66];  // [kv][d], stride 66 (pad)
  __shared__ __align__(16) _Float16 ldsV[2][64 * 66];  // [d][kv], stride 66 (pad)
  __shared__ __align__(16) _Float16 ldsP[8 * 16 * 66]; // per-wave P [m][kv]

  const int tid  = threadIdx.x;
  const int w    = tid >> 5;
  const int lane = tid & 31;
  const int hi   = lane >> 4;
  const int ln   = lane & 15;
  const int bh   = blockIdx.y;                // b*NH + h
  const int qrow0 = blockIdx.x * 128 + w * 16;
  const size_t headOff = (size_t)bh * SS * HD;
  const _Float16* kHead  = k  + headOff;
  const _Float16* vtHead = vt + headOff;      // same size per head

  // Q A-fragments (d 0..31 / 32..63), register-resident for the whole kernel
  const _Float16* qrow = q + headOff + (size_t)(qrow0 + ln) * HD;
  HFrag qa0, qa1;
#pragma unroll
  for (int p = 0; p < 8; ++p) {
    qa0.u[p] = *(const unsigned int*)(qrow + kpair(p, hi));
    qa1.u[p] = *(const unsigned int*)(qrow + 32 + kpair(p, hi));
  }

  v8f zero = {0.f,0.f,0.f,0.f,0.f,0.f,0.f,0.f};
  CAcc o[4];
#pragma unroll
  for (int t = 0; t < 4; ++t) o[t].v = zero;
  float mr[8], lr[8];
#pragma unroll
  for (int r = 0; r < 8; ++r) { mr[r] = -1e30f; lr[r] = 0.f; }

  auto issueKV = [&](int buf, int kvb) {
    // K tile: 64 kv-rows x 64 d, rows padded 32 dwords + 1 (interval code 4)
    tdm_load_2d((unsigned int)(size_t)&ldsK[buf][0],
                kHead + (size_t)kvb * HD,
                HD, SS, HD, 64, HD, 4, 0);
    // V tile: 64 d-rows x 64 kv from [B,H,D,S] layout
    tdm_load_2d((unsigned int)(size_t)&ldsV[buf][0],
                vtHead + kvb,
                SS, HD, 64, HD, SS, 4, 0);
  };

  if (w == 0) {
    issueKV(0, 0);
    __builtin_amdgcn_s_wait_tensorcnt(0);
  }
  __syncthreads();

  for (int j = 0; j < SS / 64; ++j) {
    const int cur = j & 1, nxt = cur ^ 1;
    const bool hasNext = (j + 1) < SS / 64;
    if (w == 0 && hasNext) issueKV(nxt, (j + 1) * 64);

    // S = Q K^T  (B-frags straight out of row-major K rows)
    CAcc s[4];
#pragma unroll
    for (int t = 0; t < 4; ++t) {
      HFrag b0, b1;
      const _Float16* krow = &ldsK[cur][(t * 16 + ln) * 66];
#pragma unroll
      for (int p = 0; p < 8; ++p) {
        b0.u[p] = *(const unsigned int*)(krow + kpair(p, hi));
        b1.u[p] = *(const unsigned int*)(krow + 32 + kpair(p, hi));
      }
      s[t].v = __builtin_amdgcn_wmma_f32_16x16x32_f16(false, qa0.v, false, b0.v, (short)0, zero,   false, false);
      s[t].v = __builtin_amdgcn_wmma_f32_16x16x32_f16(false, qa1.v, false, b1.v, (short)0, s[t].v, false, false);
    }

    // Online softmax (row = r + 8*hi; reduce across the 16-lane group)
    float mnew[8], alpha[8], rs[8];
#pragma unroll
    for (int r = 0; r < 8; ++r) {
      float tm = fmaxf(fmaxf(s[0].f[r], s[1].f[r]), fmaxf(s[2].f[r], s[3].f[r]));
      tm = fmaxf(tm, __shfl_xor(tm, 1));
      tm = fmaxf(tm, __shfl_xor(tm, 2));
      tm = fmaxf(tm, __shfl_xor(tm, 4));
      tm = fmaxf(tm, __shfl_xor(tm, 8));
      mnew[r]  = fmaxf(mr[r], tm);
      alpha[r] = __expf(mr[r] - mnew[r]);
      rs[r] = 0.f;
    }
#pragma unroll
    for (int t = 0; t < 4; ++t)
#pragma unroll
      for (int r = 0; r < 8; ++r) {
        float pv = __expf(s[t].f[r] - mnew[r]);
        rs[r] += pv;
        ldsP[w * 1056 + (r + (hi << 3)) * 66 + t * 16 + ln] = (_Float16)pv;
      }
#pragma unroll
    for (int r = 0; r < 8; ++r) {
      float sum = rs[r];
      sum += __shfl_xor(sum, 1);
      sum += __shfl_xor(sum, 2);
      sum += __shfl_xor(sum, 4);
      sum += __shfl_xor(sum, 8);
      lr[r] = lr[r] * alpha[r] + sum;
      mr[r] = mnew[r];
    }
#pragma unroll
    for (int t = 0; t < 4; ++t)
#pragma unroll
      for (int r = 0; r < 8; ++r) o[t].f[r] *= alpha[r];

    // O += P @ V  (P re-read in A-fragment layout; V already [d][kv])
    HFrag pa0, pa1;
    const _Float16* prow = &ldsP[w * 1056 + ln * 66];
#pragma unroll
    for (int p = 0; p < 8; ++p) {
      pa0.u[p] = *(const unsigned int*)(prow + kpair(p, hi));
      pa1.u[p] = *(const unsigned int*)(prow + 32 + kpair(p, hi));
    }
#pragma unroll
    for (int t = 0; t < 4; ++t) {
      HFrag vb0, vb1;
      const _Float16* vrow = &ldsV[cur][(t * 16 + ln) * 66];
#pragma unroll
      for (int p = 0; p < 8; ++p) {
        vb0.u[p] = *(const unsigned int*)(vrow + kpair(p, hi));
        vb1.u[p] = *(const unsigned int*)(vrow + 32 + kpair(p, hi));
      }
      o[t].v = __builtin_amdgcn_wmma_f32_16x16x32_f16(false, pa0.v, false, vb0.v, (short)0, o[t].v, false, false);
      o[t].v = __builtin_amdgcn_wmma_f32_16x16x32_f16(false, pa1.v, false, vb1.v, (short)0, o[t].v, false, false);
    }

    if (w == 0 && hasNext) __builtin_amdgcn_s_wait_tensorcnt(0);
    __syncthreads();
  }

  // Epilogue: O / l, store [B, S, H*D] fp16
  const int b = bh >> 3, h = bh & 7;
#pragma unroll
  for (int t = 0; t < 4; ++t)
#pragma unroll
    for (int r = 0; r < 8; ++r) {
      int srow = qrow0 + r + (hi << 3);
      int col  = h * HD + t * 16 + ln;
      float ov = o[t].f[r] / lr[r];
      o_out[((size_t)(b * SS + srow)) * (NH * HD) + col] = (_Float16)ov;
    }
}

extern "C" void kernel_launch(void* const* d_in, const int* in_sizes, int n_in,
                              void* d_out, int out_size, void* d_ws, size_t ws_size,
                              hipStream_t stream) {
  const float* x      = (const float*)d_in[0];   // [4,2048,512]
  const float* W_qkv  = (const float*)d_in[1];   // [512,1536]
  const float* b_qkv  = (const float*)d_in[2];   // [1536]
  const float* W_proj = (const float*)d_in[3];   // [512,512]
  const float* b_proj = (const float*)d_in[4];   // [512]
  float* out = (float*)d_out;                    // [4,2048,512]

  // fp16 workspace layout
  _Float16* xb     = (_Float16*)d_ws;                      // 8192*512
  _Float16* wqkvT  = xb    + (size_t)MTOT * EMB;           // [1536][512]
  _Float16* wprojT = wqkvT + (size_t)EMB * NQKV;           // [512][512]
  _Float16* qkv    = wprojT + (size_t)EMB * EMB;           // q | k | v_t
  _Float16* attn   = qkv   + 3 * QSZ;                      // 8192*512

  int nx = MTOT * EMB;
  cvt_f32_to_f16<<<(nx + 255) / 256, 256, 0, stream>>>(x, xb, nx);
  int nwq = EMB * NQKV;
  cvt_transpose_f32_to_f16<<<(nwq + 255) / 256, 256, 0, stream>>>(W_qkv, wqkvT, EMB, NQKV);
  int nwp = EMB * EMB;
  cvt_transpose_f32_to_f16<<<(nwp + 255) / 256, 256, 0, stream>>>(W_proj, wprojT, EMB, EMB);

  // QKV projection: [8192,512] x [512,1536] -> scatter q/k (seq-major), v (d-major)
  gemm_f16_wmma<<<dim3(NQKV / 128, MTOT / 128), 256, 0, stream>>>(
      xb, wqkvT, b_qkv, EMB, NQKV, 0, qkv, nullptr);

  // Fused flash attention per (b,h), 128-row q tiles
  flash_attn_wmma<<<dim3(SS / 128, BB * NH), 256, 0, stream>>>(
      qkv, qkv + QSZ, qkv + 2 * QSZ, attn);

  // Output projection: [8192,512] x [512,512] -> f32 out
  gemm_f16_wmma<<<dim3(EMB / 128, MTOT / 128), 256, 0, stream>>>(
      attn, wprojT, b_proj, EMB, EMB, 1, nullptr, out);
}
